// ConvLSTMCellSpike_58514634440960
// MI455X (gfx1250) — matrix operand order repaired and checked by reading, not compiled, not run on hardware
//
#include <hip/hip_runtime.h>
#include <hip/hip_bf16.h>

typedef __attribute__((ext_vector_type(16))) _Float16 v16h;
typedef __attribute__((ext_vector_type(8)))  float    v8f;

#define DTM 0.1f   // DT * TAU_MEM_INV
#define DECAY 0.8f // 1 - DT * TAU_SYN_INV
#define NCH 192    // live output channels (o-gate conv columns are dead)
#define NELEM 524288   // B*C*H*W = 8*64*32*32
#define NPOS  8192     // B*H*W

// ---------------- weight / input prep ----------------

// weight_ih (only co<192 used) -> f16 [K=576][192], k = (kh*3+kw)*64 + ci
__global__ void k_prep_wih(const float* __restrict__ wih, _Float16* __restrict__ out) {
  int idx = blockIdx.x * blockDim.x + threadIdx.x;
  if (idx >= 576 * NCH) return;
  int k = idx / NCH, co = idx % NCH;
  int p = k >> 6, ci = k & 63;
  out[idx] = (_Float16)wih[(co * 64 + ci) * 9 + p];
}

// combined hidden weights -> f16 [K=1152][192], k = (kh*3+kw)*128 + c
__global__ void k_prep_whcw(const float* __restrict__ whh, const float* __restrict__ wch,
                            _Float16* __restrict__ out) {
  int idx = blockIdx.x * blockDim.x + threadIdx.x;
  if (idx >= 1152 * NCH) return;
  int k = idx / NCH, co = idx % NCH;
  int p = k >> 7, c = k & 127;
  float v;
  if (c < 64)        v = whh[(co * 64 + c) * 9 + p];
  else if (co < 128) v = wch[(co * 64 + (c - 64)) * 9 + p];
  else               v = 0.f;   // g gate has no peephole
  out[idx] = (_Float16)v;
}

// combined bias: bias_ih + bias_hh + peephole bias_ch (i,f only)
__global__ void k_prep_bias(const float* __restrict__ bih, const float* __restrict__ bhh,
                            const float* __restrict__ bch, float* __restrict__ out) {
  int co = threadIdx.x;
  if (co >= NCH) return;
  float pb = (co < 128) ? bch[co] : 0.f;
  out[co] = bih[co] + bhh[co] + pb;
}

// stack h0||c0 into f16 [B][128][32][32]
__global__ void k_prep_hc(const float* __restrict__ h0, const float* __restrict__ c0,
                          _Float16* __restrict__ out) {
  int idx = blockIdx.x * blockDim.x + threadIdx.x;
  if (idx >= 1048576) return;
  int b = idx >> 17, rem = idx & 131071, c = rem >> 10, hw = rem & 1023;
  float v = (c < 64) ? h0[((b << 6) + c) * 1024 + hw]
                     : c0[((b << 6) + (c - 64)) * 1024 + hw];
  out[idx] = (_Float16)v;
}

__global__ void k_init_state(float* venc, float* vi, float* ii, float* vf, float* fi,
                             float* vg, float* gi, float* mi, float* mf, float* mg, float* mo) {
  int idx = blockIdx.x * blockDim.x + threadIdx.x;
  if (idx >= NELEM) return;
  venc[idx] = 0.f; vi[idx] = 0.f; ii[idx] = 0.f; vf[idx] = 0.f; fi[idx] = 0.f;
  vg[idx] = 0.f; gi[idx] = 0.f;
  float ninf = __int_as_float(0xff800000);
  mi[idx] = ninf; mf[idx] = ninf; mg[idx] = ninf; mo[idx] = ninf;
}

// ---------------- per-step elementwise kernels ----------------

__global__ void k_encoder(const float* __restrict__ inp, float* __restrict__ venc,
                          _Float16* __restrict__ x16) {
  int idx = blockIdx.x * blockDim.x + threadIdx.x;
  if (idx >= NELEM) return;
  float v0 = venc[idx];
  float v = v0 + DTM * (inp[idx] - v0);
  float x = (v - 1.0f > 0.0f) ? 1.0f : 0.0f;
  venc[idx] = v - x * v;
  x16[idx] = (_Float16)x;
}

__global__ void k_finalize(const float* __restrict__ c0,
                           const float* __restrict__ mi, const float* __restrict__ mf,
                           const float* __restrict__ mg, const float* __restrict__ mo,
                           float* __restrict__ out) {
  int idx = blockIdx.x * blockDim.x + threadIdx.x;
  if (idx >= NELEM) return;
  float c1 = mf[idx] * c0[idx] + mi[idx] * mg[idx];
  out[idx] = mo[idx] * tanhf(c1);     // h1
  out[NELEM + idx] = c1;              // c1
}

// ---------------- generic implicit-GEMM conv (one-time hidden conv) ----------------
__global__ __launch_bounds__(128) void k_conv_wmma(
    const _Float16* __restrict__ X,   // [B][Cin][32][32] f16
    const _Float16* __restrict__ Wt,  // [9*Cin][NCH] f16
    const float* __restrict__ Cbias,  // [NCH] per-channel init
    float* __restrict__ Out,          // [B][NCH][32][32]
    int cinShift, int nKc) {
  __shared__ _Float16 As[64][34];
  __shared__ _Float16 Bs[32][68];
  const int tid = threadIdx.x;
  const int lane = tid & 31, wv = tid >> 5;
  const int half = lane >> 4, l16 = lane & 15;
  const int tileM = blockIdx.x << 6;
  const int tileN = blockIdx.y << 6;
  const int cmask = (1 << cinShift) - 1;

  v8f acc[4];
#pragma unroll
  for (int nn = 0; nn < 4; ++nn) {
    float cv = Cbias[tileN + (nn << 4) + l16];
#pragma unroll
    for (int r = 0; r < 8; ++r) acc[nn][r] = cv;
  }

  for (int kc = 0; kc < nKc; ++kc) {
#pragma unroll
    for (int e = tid; e < 2048; e += 128) {
      int i = e >> 5, kk = e & 31;
      int m = tileM + i;
      int b = m >> 10, hw = m & 1023, h = hw >> 5, w = hw & 31;
      int k = (kc << 5) + kk;
      int p = k >> cinShift, ci = k & cmask;
      int hh = h + p / 3 - 1, ww = w + p % 3 - 1;
      _Float16 v = (_Float16)0.f;
      if ((unsigned)hh < 32u && (unsigned)ww < 32u)
        v = X[(((b << cinShift) + ci) << 10) + (hh << 5) + ww];
      As[i][kk] = v;
    }
#pragma unroll
    for (int e = tid; e < 2048; e += 128) {
      int kk = e >> 6, n = e & 63;
      Bs[kk][n] = Wt[((kc << 5) + kk) * NCH + tileN + n];
    }
    __syncthreads();

    v16h a;
    int arow = (wv << 4) + l16;
#pragma unroll
    for (int r = 0; r < 8; ++r) {
      int kb = ((r >> 2) << 4) + (half << 3) + ((r & 3) << 1);
      a[2 * r]     = As[arow][kb];
      a[2 * r + 1] = As[arow][kb + 1];
    }
#pragma unroll
    for (int nn = 0; nn < 4; ++nn) {
      v16h bfr;
      int ncol = (nn << 4) + l16;
#pragma unroll
      for (int r = 0; r < 8; ++r) {
        int kb = ((r >> 2) << 4) + (half << 3) + ((r & 3) << 1);
        bfr[2 * r]     = Bs[kb][ncol];
        bfr[2 * r + 1] = Bs[kb + 1][ncol];
      }
      acc[nn] = __builtin_amdgcn_wmma_f32_16x16x32_f16(
          false, a, false, bfr, (short)0, acc[nn], false, false);
    }
    __syncthreads();
  }

#pragma unroll
  for (int nn = 0; nn < 4; ++nn) {
    int co = tileN + (nn << 4) + l16;
#pragma unroll
    for (int r = 0; r < 8; ++r) {
      int m = tileM + (wv << 4) + r + (half << 3);
      int b = m >> 10, hw = m & 1023;
      Out[((b * NCH + co) << 10) + hw] = acc[nn][r];
    }
  }
}

// ---------------- fused per-step conv + LIF gate update ----------------
// One workgroup (256 thr = 8 wave32) owns 64 positions x ALL 192 channels:
// wave w -> M subtile (w&3), N cols [(w>>2)*96, +96) as six 16-wide subtiles.
// After the K loop the 64x192 f32 result is exchanged via LDS [chan][pos]
// (pitch 65 -> conflict-free both directions) and the 4-gate LIF update +
// running maxima run in-block. Eliminates the wxhc HBM round-trip and keeps
// the g->o dependency (o-gate reads the just-updated g state) local.
__global__ __launch_bounds__(256) void k_conv_gates_wmma(
    const _Float16* __restrict__ X,   // spikes f16 [B][64][32][32]
    const _Float16* __restrict__ Wt,  // [576][192] f16
    const float* __restrict__ whc,    // [B][192][32][32] C-init (incl. all biases)
    float* vi, float* ii, float* vf, float* fi, float* vg, float* gi,
    float* mi, float* mf, float* mg, float* mo, int upd) {
  __shared__ union {
    struct {
      _Float16 As[64][34];
      _Float16 Bs[32][200];
    } st;
    float Ds[192][65];
  } u;

  const int tid = threadIdx.x;
  const int lane = tid & 31, wv = tid >> 5;
  const int half = lane >> 4, l16 = lane & 15;
  const int tileM = blockIdx.x << 6;
  const int b = tileM >> 10, hw0 = tileM & 1023;
  const int mrb = (wv & 3) << 4;
  const int ncb = (wv >> 2) * 96;

  v8f acc[6];
#pragma unroll
  for (int s = 0; s < 6; ++s) {
    int co = ncb + (s << 4) + l16;
#pragma unroll
    for (int r = 0; r < 8; ++r) {
      int hw = hw0 + mrb + r + (half << 3);
      acc[s][r] = whc[((b * NCH + co) << 10) + hw];
    }
  }

  for (int kc = 0; kc < 18; ++kc) {
#pragma unroll
    for (int e = tid; e < 2048; e += 256) {
      int i = e >> 5, kk = e & 31;
      int hw = hw0 + i;
      int h = hw >> 5, w = hw & 31;
      int k = (kc << 5) + kk;
      int p = k >> 6, ci = k & 63;
      int hh = h + p / 3 - 1, ww = w + p % 3 - 1;
      _Float16 v = (_Float16)0.f;
      if ((unsigned)hh < 32u && (unsigned)ww < 32u)
        v = X[(((b << 6) + ci) << 10) + (hh << 5) + ww];
      u.st.As[i][kk] = v;
    }
#pragma unroll
    for (int e = tid; e < 6144; e += 256) {
      int kk = e / 192, n = e - kk * 192;
      u.st.Bs[kk][n] = Wt[((kc << 5) + kk) * NCH + n];
    }
    __syncthreads();

    v16h a;
    int arow = mrb + l16;
#pragma unroll
    for (int r = 0; r < 8; ++r) {
      int kb = ((r >> 2) << 4) + (half << 3) + ((r & 3) << 1);
      a[2 * r]     = u.st.As[arow][kb];
      a[2 * r + 1] = u.st.As[arow][kb + 1];
    }
#pragma unroll
    for (int s = 0; s < 6; ++s) {
      v16h bfr;
      int ncol = ncb + (s << 4) + l16;
#pragma unroll
      for (int r = 0; r < 8; ++r) {
        int kb = ((r >> 2) << 4) + (half << 3) + ((r & 3) << 1);
        bfr[2 * r]     = u.st.Bs[kb][ncol];
        bfr[2 * r + 1] = u.st.Bs[kb + 1][ncol];
      }
      acc[s] = __builtin_amdgcn_wmma_f32_16x16x32_f16(
          false, a, false, bfr, (short)0, acc[s], false, false);
    }
    __syncthreads();
  }

  // exchange D through LDS as [channel][position]
#pragma unroll
  for (int s = 0; s < 6; ++s) {
    int co = ncb + (s << 4) + l16;
#pragma unroll
    for (int r = 0; r < 8; ++r)
      u.Ds[co][mrb + r + (half << 3)] = acc[s][r];
  }
  __syncthreads();

  // fused LIF updates: 64 state channels x 64 positions
#pragma unroll
  for (int e = tid; e < 4096; e += 256) {
    int c = e >> 6, pos = e & 63;
    float wi = u.Ds[c][pos];
    float wf = u.Ds[c + 64][pos];
    float wg = u.Ds[c + 128][pos];
    int idx = (b << 16) + (c << 10) + hw0 + pos;

    float v, cur, vd, z, vn;
    // i gate
    v = vi[idx]; cur = ii[idx];
    vd = v + DTM * (cur - v);
    z = (vd - 1.f > 0.f) ? 1.f : 0.f;
    vn = (1.f - z) * vd;
    vi[idx] = vn; ii[idx] = DECAY * cur + wi;
    if (upd) mi[idx] = fmaxf(mi[idx], vn);
    // f gate
    v = vf[idx]; cur = fi[idx];
    vd = v + DTM * (cur - v);
    z = (vd - 1.f > 0.f) ? 1.f : 0.f;
    vn = (1.f - z) * vd;
    vf[idx] = vn; fi[idx] = DECAY * cur + wf;
    if (upd) mf[idx] = fmaxf(mf[idx], vn);
    // g gate
    v = vg[idx]; cur = gi[idx];
    vd = v + DTM * (cur - v);
    z = (vd - 1.f > 0.f) ? 1.f : 0.f;
    float vgn = (1.f - z) * vd;
    float gin = DECAY * cur + wg;
    vg[idx] = vgn; gi[idx] = gin;
    if (upd) mg[idx] = fmaxf(mg[idx], vgn);
    // o gate: uses just-updated g state; its own input current is discarded
    float vdo = vgn + DTM * (gin - vgn);
    float zo = (vdo - 1.f > 0.f) ? 1.f : 0.f;
    float von = (1.f - zo) * vdo;
    if (upd) mo[idx] = fmaxf(mo[idx], von);
  }
}

// ---------------- launch ----------------
extern "C" void kernel_launch(void* const* d_in, const int* in_sizes, int n_in,
                              void* d_out, int out_size, void* d_ws, size_t ws_size,
                              hipStream_t stream) {
  const float* input = (const float*)d_in[0];
  const float* h0    = (const float*)d_in[1];
  const float* c0    = (const float*)d_in[2];
  const float* w_ih  = (const float*)d_in[3];
  const float* w_hh  = (const float*)d_in[4];
  const float* w_ch  = (const float*)d_in[5];
  const float* b_ih  = (const float*)d_in[6];
  const float* b_hh  = (const float*)d_in[7];
  const float* b_ch  = (const float*)d_in[8];
  float* out = (float*)d_out;

  char* ws = (char*)d_ws;
  size_t off = 0;
  auto take = [&](size_t bytes) {
    char* p = ws + off;
    off = (off + bytes + 255) & ~(size_t)255;
    return (void*)p;
  };
  _Float16* x16    = (_Float16*)take((size_t)NELEM * 2);
  _Float16* hc16   = (_Float16*)take((size_t)1048576 * 2);
  _Float16* wih16  = (_Float16*)take((size_t)576 * NCH * 2);
  _Float16* whcw16 = (_Float16*)take((size_t)1152 * NCH * 2);
  float* biasC = (float*)take((size_t)NCH * 4);
  float* whc   = (float*)take((size_t)8 * NCH * 1024 * 4);
  float* venc = (float*)take((size_t)NELEM * 4);
  float* vi = (float*)take((size_t)NELEM * 4);
  float* ii = (float*)take((size_t)NELEM * 4);
  float* vf = (float*)take((size_t)NELEM * 4);
  float* fi = (float*)take((size_t)NELEM * 4);
  float* vg = (float*)take((size_t)NELEM * 4);
  float* gi = (float*)take((size_t)NELEM * 4);
  float* mi = (float*)take((size_t)NELEM * 4);
  float* mf = (float*)take((size_t)NELEM * 4);
  float* mg = (float*)take((size_t)NELEM * 4);
  float* mo = (float*)take((size_t)NELEM * 4);

  // prep
  k_prep_wih<<<(576 * NCH + 255) / 256, 256, 0, stream>>>(w_ih, wih16);
  k_prep_whcw<<<(1152 * NCH + 255) / 256, 256, 0, stream>>>(w_hh, w_ch, whcw16);
  k_prep_bias<<<1, 256, 0, stream>>>(b_ih, b_hh, b_ch, biasC);
  k_prep_hc<<<1048576 / 256, 256, 0, stream>>>(h0, c0, hc16);
  k_init_state<<<NELEM / 256, 256, 0, stream>>>(venc, vi, ii, vf, fi, vg, gi, mi, mf, mg, mo);

  // one-time hidden conv: whc = conv(h0||c0) + all biases, K = 1152
  k_conv_wmma<<<dim3(NPOS / 64, NCH / 64), 128, 0, stream>>>(
      hc16, whcw16, biasC, whc, 7, 36);

  // 32-step scan: encoder + fused conv/gates
  for (int t = 0; t < 32; ++t) {
    k_encoder<<<NELEM / 256, 256, 0, stream>>>(input, venc, x16);
    k_conv_gates_wmma<<<NPOS / 64, 256, 0, stream>>>(
        x16, wih16, whc, vi, ii, vf, fi, vg, gi, mi, mf, mg, mo, (t >= 1) ? 1 : 0);
  }

  k_finalize<<<NELEM / 256, 256, 0, stream>>>(c0, mi, mf, mg, mo, out);
}
